// DCNv3_pytorch_3d_39625368273204
// MI455X (gfx1250) — matrix operand order repaired and checked by reading, not compiled, not run on hardware
//
#include <hip/hip_runtime.h>
#include <hip/hip_bf16.h>
#include <math.h>

// ---------------------------------------------------------------------------
// DCNv3-3D forward for MI455X (gfx1250, wave32, WMMA, async-LDS).
//   k_pack       : one-time pack of all weight matrices into f16 WMMA
//                  B-fragment layout (72 x 1KB fragments in ws)
//   k_in_proj    : x  = input @ in_w^T + in_b            (WMMA f32<-f16)
//   k_dw_ln_gelu : x1 = GELU(LN(dwconv3x3x3(input)))     -> f16 (A-operand)
//   k_core       : async-copy x1 tile to LDS -> head GEMM (WMMA) -> softmax
//                  -> deformable sampling -> out-proj (WMMA) -> d_out
// ---------------------------------------------------------------------------

typedef __attribute__((ext_vector_type(16))) _Float16 v16h;
typedef __attribute__((ext_vector_type(8)))  float    v8f;

#define CC    64          // channels
#define GG    4           // groups
#define PP    27          // kernel points
#define NN    2
#define DD    16
#define HHH   32
#define WWW   32
#define MTOT  (NN*DD*HHH*WWW)   // 32768 voxels
#define OFFD  (GG*PP*3)         // 324
#define MSKD  (GG*PP)           // 108
#define HEADW 448               // 28 tiles of 16 (324+108 padded)
#define NFRAG 72                // 8 (in_w) + 8 (out_w) + 56 (head)

#define WMMA_F32_F16(A,B,Cacc) \
  __builtin_amdgcn_wmma_f32_16x16x32_f16(false,(A),false,(B),(short)0,(Cacc),false,false)

// Async global->LDS b128 copy + wait (gfx1250 ASYNCcnt path).
#define ASYNC_CP_B128(ldsOff, gptr)                                        \
  asm volatile("global_load_async_to_lds_b128 %0, %1, off"                 \
               :: "v"(ldsOff), "v"(gptr) : "memory")
#define ASYNC_WAIT0() asm volatile("s_wait_asynccnt 0" ::: "memory")

// A fragment (16x32 f16) from row-major f32 (with convert).
// lanes 0-15: halves 0..7 -> K 0..7, halves 8..15 -> K 16..23
// lanes16-31: halves 0..7 -> K 8..15, halves 8..15 -> K 24..31
static __device__ inline v16h load_a_f32(const float* src, int ld, int row0,
                                         int kk, int lane) {
  int mrow  = row0 + (lane & 15);
  int kbase = kk + ((lane >> 4) & 1) * 8;
  const float* p = src + (size_t)mrow * ld + kbase;
  v16h a;
#pragma unroll
  for (int h = 0; h < 8; ++h) {
    a[h]     = (_Float16)p[h];
    a[h + 8] = (_Float16)p[h + 16];
  }
  return a;
}

// A fragment from row-major f16 (two 16B vector loads per lane).
static __device__ inline v16h load_a_f16(const _Float16* src, int ld, int row0,
                                         int kk, int lane) {
  int mrow  = row0 + (lane & 15);
  int kbase = kk + ((lane >> 4) & 1) * 8;
  const _Float16* p = src + (size_t)mrow * ld + kbase;
  v16h a;
#pragma unroll
  for (int h = 0; h < 8; ++h) {
    a[h]     = p[h];
    a[h + 8] = p[h + 16];
  }
  return a;
}

// trilinear corner in padded coords (Din=18,Hin=34,Win=34); padding ring and
// out-of-range are zero, so one combined validity test suffices.
static __device__ inline float cornerv(const float* __restrict__ xv, int nb,
                                       int zp, int yp, int xp, int c) {
  bool ok = ((unsigned)(xp - 1) < 32u) & ((unsigned)(yp - 1) < 32u) &
            ((unsigned)(zp - 1) < 16u);
  if (!ok) return 0.0f;
  size_t m = (((size_t)nb * 16 + (zp - 1)) * 32 + (yp - 1)) * 32 + (xp - 1);
  return xv[m * CC + c];
}

// ------------------ kernel 0: pack weights into B fragments -----------------
// Fragment f, lane l, half h  maps to  B[k][n] = W[n][k] with
//   n = col0 + (l&15),  k = (f&1)*32 + (l>>4)*8 + (h&7) + (h>>3)*16.
// frag 0..7: in_w tiles; 8..15: out_w tiles; 16..71: [off_w;mask_w;0] tiles.
__global__ void __launch_bounds__(256)
k_pack(const float* __restrict__ wIn, const float* __restrict__ wOut,
       const float* __restrict__ offw, const float* __restrict__ mskw,
       _Float16* __restrict__ bf) {
  int idx = blockIdx.x * 256 + threadIdx.x;      // (frag, lane)
  if (idx >= NFRAG * 32) return;
  int frag = idx >> 5, lane = idx & 31;
  int nloc = lane & 15;
  int kk   = (frag & 1) * 32 + ((lane >> 4) & 1) * 8;
  const float* W = nullptr; int n = 0;
  if (frag < 8)        { W = wIn;  n = (frag >> 1) * 16 + nloc; }
  else if (frag < 16)  { W = wOut; n = ((frag - 8) >> 1) * 16 + nloc; }
  else {
    int tcol = ((frag - 16) >> 1) * 16 + nloc;   // 0..447
    if (tcol < OFFD)             { W = offw; n = tcol; }
    else if (tcol < OFFD + MSKD) { W = mskw; n = tcol - OFFD; }
  }
  _Float16* dst = bf + ((size_t)frag * 32 + lane) * 16;
#pragma unroll
  for (int h = 0; h < 16; ++h)
    dst[h] = W ? (_Float16)W[(size_t)n * CC + kk + (h & 7) + (h >> 3) * 16] : (_Float16)0.0f;
}

// ------------------------- kernel 1: input projection -----------------------
__global__ void __launch_bounds__(128)
k_in_proj(const float* __restrict__ in, const _Float16* __restrict__ bf,
          const float* __restrict__ bIn, float* __restrict__ xo) {
  const v16h* bfv = (const v16h*)bf;
  int lane = threadIdx.x & 31;
  int wave = threadIdx.x >> 5;           // 4 waves = 4 col tiles
  int row0 = blockIdx.x * 16;
  v8f acc = {};
  v16h a = load_a_f32(in, CC, row0, 0, lane);
  v16h b = bfv[(wave * 2 + 0) * 32 + lane];
  acc = WMMA_F32_F16(a, b, acc);
  a = load_a_f32(in, CC, row0, 32, lane);
  b = bfv[(wave * 2 + 1) * 32 + lane];
  acc = WMMA_F32_F16(a, b, acc);
  int nc = wave * 16 + (lane & 15);
  float bv = bIn[nc];
  int mb = row0 + 8 * (lane >> 4);
#pragma unroll
  for (int r = 0; r < 8; ++r)
    xo[(size_t)(mb + r) * CC + nc] = acc[r] + bv;
}

// -------------- kernel 2: dwconv + LayerNorm + GELU -> f16 x1 ---------------
__global__ void __launch_bounds__(256)
k_dw_ln_gelu(const float* __restrict__ in, const float* __restrict__ dww,
             const float* __restrict__ dwb, const float* __restrict__ lng,
             const float* __restrict__ lnb, _Float16* __restrict__ x1) {
  __shared__ float vals[4][CC];
  __shared__ float stats[4][2];
  int t  = threadIdx.x;
  int vl = t >> 6;                       // voxel in block (4 per block)
  int c  = t & 63;
  int m  = blockIdx.x * 4 + vl;
  int w = m & 31, h = (m >> 5) & 31, d = (m >> 10) & 15, n = m >> 14;
  float acc = dwb[c];
  const float* wp = dww + c * 27;
#pragma unroll
  for (int kd = 0; kd < 3; ++kd) {
    int dz = d + kd - 1;
#pragma unroll
    for (int kh = 0; kh < 3; ++kh) {
      int hy = h + kh - 1;
#pragma unroll
      for (int kw = 0; kw < 3; ++kw) {
        int wx = w + kw - 1;
        float v = 0.0f;
        if (dz >= 0 && dz < 16 && hy >= 0 && hy < 32 && wx >= 0 && wx < 32)
          v = in[(size_t)((((n * 16 + dz) * 32 + hy) * 32 + wx)) * CC + c];
        acc = fmaf(v, wp[(kd * 3 + kh) * 3 + kw], acc);
      }
    }
  }
  vals[vl][c] = acc;
  __syncthreads();
  if (c == 0) {
    float s = 0.0f, s2 = 0.0f;
    for (int i = 0; i < CC; ++i) { float v = vals[vl][i]; s += v; s2 += v * v; }
    float mu  = s * (1.0f / CC);
    float var = s2 * (1.0f / CC) - mu * mu;
    stats[vl][0] = mu;
    stats[vl][1] = rsqrtf(var + 1e-6f);
  }
  __syncthreads();
  float xn = (acc - stats[vl][0]) * stats[vl][1] * lng[c] + lnb[c];
  float ge = 0.5f * xn * (1.0f + erff(xn * 0.70710678118654752f));
  x1[(size_t)m * CC + c] = (_Float16)ge;
}

// ------ kernel 3: fused heads + softmax + deformable sampling + out-proj ----
__global__ void __launch_bounds__(256)
k_core(const _Float16* __restrict__ x1, const float* __restrict__ xv,
       const _Float16* __restrict__ bf, const float* __restrict__ offb,
       const float* __restrict__ mskb, const float* __restrict__ outb,
       float* __restrict__ out) {
  __shared__ float head[16][HEADW];            // offsets [0,324), masks [324,432)
  __shared__ float posx[16 * GG * PP];
  __shared__ float posy[16 * GG * PP];
  __shared__ float posz[16 * GG * PP];
  __shared__ _Float16 x1t[16 * CC];            // 2KB async-staged x1 tile
  __shared__ _Float16 otile[16 * (CC + 8)];    // f16 out tile (A-operand)

  const v16h* bfv = (const v16h*)bf;
  int t    = threadIdx.x;
  int lane = t & 31;
  int wave = t >> 5;                           // 8 waves
  int row0 = blockIdx.x * 16;

  __builtin_prefetch(bf + 16 * 512, 0, 1);     // head fragments -> L2/WGP$

  // ---- phase 0: async stage x1 tile (16x64 f16 = 2KB) into LDS -------------
  if (t < 128) {
    unsigned lofs = (unsigned)(size_t)(&x1t[0]) + (unsigned)t * 16u;
    const char* g = (const char*)(x1 + (size_t)row0 * CC) + (size_t)t * 16u;
    ASYNC_CP_B128(lofs, g);
    ASYNC_WAIT0();
  }
  __syncthreads();

  // ---- phase 1: head GEMM, 28 col tiles over 8 waves (wave-uniform guard) --
  v16h a0 = load_a_f16(&x1t[0], CC, 0, 0, lane);
  v16h a1 = load_a_f16(&x1t[0], CC, 0, 32, lane);
#pragma unroll
  for (int i = 0; i < 4; ++i) {
    int tt = wave + 8 * i;
    if (tt < 28) {                             // uniform per wave -> EXEC full
      v8f acc = {};
      v16h b = bfv[(16 + tt * 2 + 0) * 32 + lane];
      acc = WMMA_F32_F16(a0, b, acc);
      b = bfv[(16 + tt * 2 + 1) * 32 + lane];
      acc = WMMA_F32_F16(a1, b, acc);
      int nc = tt * 16 + (lane & 15);
      float bv = (nc < OFFD) ? offb[nc]
               : ((nc < OFFD + MSKD) ? mskb[nc - OFFD] : 0.0f);
      int mb = 8 * (lane >> 4);
#pragma unroll
      for (int r = 0; r < 8; ++r) head[mb + r][nc] = acc[r] + bv;
    }
  }
  __syncthreads();

  // ---- phase 2: sample positions (padded-grid coords) + mask softmax -------
  // grid-sample math collapses to: px = w + i + off_x (padded Win=34), etc.
  for (int task = t; task < 16 * GG * PP; task += 256) {
    int p   = task % PP;
    int g   = (task / PP) & 3;
    int row = task / (PP * GG);
    int m = row0 + row;
    int w = m & 31, h = (m >> 5) & 31, d = (m >> 10) & 15;
    int i = p / 9, j = (p / 3) % 3, kz = p % 3;
    const float* o = &head[row][g * (PP * 3) + p * 3];
    posx[task] = (float)(w + i)  + o[0];
    posy[task] = (float)(h + j)  + o[1];
    posz[task] = (float)(d + kz) + o[2];
  }
  if (t < 64) {                                 // one (row, group) per thread
    int row = t >> 2, g = t & 3;
    float* ml = &head[row][OFFD + g * PP];
    float mx = ml[0];
    for (int p = 1; p < PP; ++p) mx = fmaxf(mx, ml[p]);
    float s = 0.0f;
    for (int p = 0; p < PP; ++p) { float e = expf(ml[p] - mx); ml[p] = e; s += e; }
    float inv = 1.0f / s;
    for (int p = 0; p < PP; ++p) ml[p] *= inv;
  }
  __syncthreads();

  // ---- phase 3: deformable trilinear sampling (16 rows x 64 channels) ------
#pragma unroll
  for (int it = 0; it < 4; ++it) {
    int task = t + 256 * it;                    // 1024 tasks
    int row = task >> 6;
    int c   = task & 63;
    int g   = c >> 4;
    int m   = row0 + row;
    int nb  = m >> 14;
    int pb  = (row * GG + g) * PP;
    float acc = 0.0f;
#pragma unroll 1
    for (int p = 0; p < PP; ++p) {
      float px = posx[pb + p], py = posy[pb + p], pz = posz[pb + p];
      float msk = head[row][OFFD + g * PP + p];
      float fx0 = floorf(px), fy0 = floorf(py), fz0 = floorf(pz);
      int x0 = (int)fx0, y0 = (int)fy0, z0 = (int)fz0;
      float fx = px - fx0, fy = py - fy0, fz = pz - fz0;
      float gx = 1.0f - fx, gy = 1.0f - fy, gz = 1.0f - fz;
      float v =
        cornerv(xv, nb, z0,     y0,     x0,     c) * gz * gy * gx +
        cornerv(xv, nb, z0,     y0,     x0 + 1, c) * gz * gy * fx +
        cornerv(xv, nb, z0,     y0 + 1, x0,     c) * gz * fy * gx +
        cornerv(xv, nb, z0,     y0 + 1, x0 + 1, c) * gz * fy * fx +
        cornerv(xv, nb, z0 + 1, y0,     x0,     c) * fz * gy * gx +
        cornerv(xv, nb, z0 + 1, y0,     x0 + 1, c) * fz * gy * fx +
        cornerv(xv, nb, z0 + 1, y0 + 1, x0,     c) * fz * fy * gx +
        cornerv(xv, nb, z0 + 1, y0 + 1, x0 + 1, c) * fz * fy * fx;
      acc = fmaf(msk, v, acc);
    }
    otile[row * (CC + 8) + c] = (_Float16)acc;
  }
  __syncthreads();

  // ---- phase 4: output projection (4 col tiles, waves 0..3) ----------------
  if (wave < 4) {                               // uniform per wave
    v16h oa0 = load_a_f16(&otile[0], CC + 8, 0, 0, lane);
    v16h oa1 = load_a_f16(&otile[0], CC + 8, 0, 32, lane);
    v8f acc = {};
    v16h b = bfv[(8 + wave * 2 + 0) * 32 + lane];
    acc = WMMA_F32_F16(oa0, b, acc);
    b = bfv[(8 + wave * 2 + 1) * 32 + lane];
    acc = WMMA_F32_F16(oa1, b, acc);
    int nc = wave * 16 + (lane & 15);
    float bv = outb[nc];
    int mb = 8 * (lane >> 4);
#pragma unroll
    for (int r = 0; r < 8; ++r)
      out[(size_t)(row0 + mb + r) * CC + nc] = acc[r] + bv;
  }
}

// ---------------------------------------------------------------------------
extern "C" void kernel_launch(void* const* d_in, const int* in_sizes, int n_in,
                              void* d_out, int out_size, void* d_ws, size_t ws_size,
                              hipStream_t stream) {
  const float* input = (const float*)d_in[0];
  const float* dw_w  = (const float*)d_in[1];
  const float* dw_b  = (const float*)d_in[2];
  const float* ln_g  = (const float*)d_in[3];
  const float* ln_b  = (const float*)d_in[4];
  const float* off_w = (const float*)d_in[5];
  const float* off_b = (const float*)d_in[6];
  const float* mask_w= (const float*)d_in[7];
  const float* mask_b= (const float*)d_in[8];
  const float* in_w  = (const float*)d_in[9];
  const float* in_b  = (const float*)d_in[10];
  const float* out_w = (const float*)d_in[11];
  const float* out_b = (const float*)d_in[12];

  float*    x   = (float*)d_ws;                     // 8 MB: value path (f32)
  _Float16* x1h = (_Float16*)(x + (size_t)MTOT*CC); // 4 MB: query path (f16)
  _Float16* bfr = x1h + (size_t)MTOT*CC;            // 72 KB: packed B frags
  float*    o   = (float*)d_out;

  k_pack     <<<(NFRAG*32 + 255)/256, 256, 0, stream>>>(in_w, out_w, off_w,
                                                        mask_w, bfr);
  k_in_proj  <<<MTOT / 16, 128, 0, stream>>>(input, bfr, in_b, x);
  k_dw_ln_gelu<<<MTOT / 4, 256, 0, stream>>>(input, dw_w, dw_b, ln_g, ln_b, x1h);
  k_core     <<<MTOT / 16, 256, 0, stream>>>(x1h, x, bfr, off_b, mask_b,
                                             out_b, o);
}
